// ChannelAttention_59287728554391
// MI455X (gfx1250) — matrix operand order repaired and checked by reading, not compiled, not run on hardware
//
#include <hip/hip_runtime.h>

typedef __bf16 bf16;
typedef __attribute__((ext_vector_type(8)))  __bf16 v8bf;
typedef __attribute__((ext_vector_type(16))) __bf16 v16bf;
typedef __attribute__((ext_vector_type(8)))  float  v8f;

#define NB  64
#define CC  512
#define HW  2304
#define KB  32
#define LDT 40   // padded LDS row stride in ushorts (80B -> conflict-free 16B reads)

__device__ __forceinline__ unsigned short f2bf(float f) {
  unsigned int u = __builtin_bit_cast(unsigned int, f);
  u += 0x7FFFu + ((u >> 16) & 1u);            // round-to-nearest-even
  return (unsigned short)(u >> 16);
}

// ---- CDNA5 async copy: 16 bytes per lane, global -> LDS, tracked by ASYNCcnt
__device__ __forceinline__ void async_b128(unsigned lds_addr, const unsigned short* gaddr) {
  asm volatile("global_load_async_to_lds_b128 %0, %1, off"
               :: "v"(lds_addr), "v"(gaddr) : "memory");
}
__device__ __forceinline__ void wait_async0() {
  asm volatile("s_wait_asynccnt 0x0" ::: "memory");
}
__device__ __forceinline__ unsigned lds_off(const void* p) {
  return (unsigned)(unsigned long long)p;   // low 32 bits of generic ptr = LDS byte addr
}

// A-operand fragment: 16x32 bf16. Lanes 0-15: M=0..15, K={0..7,16..23};
// lanes 16-31: same M, K={8..15,24..31}.
__device__ __forceinline__ v16bf frag_a(const unsigned short* lds, int row, int lane) {
  const int ko = (lane < 16) ? 0 : 8;
  v8bf lo = *(const v8bf*)(lds + row * LDT + ko);
  v8bf hi = *(const v8bf*)(lds + row * LDT + 16 + ko);
  return __builtin_shufflevector(lo, hi, 0,1,2,3,4,5,6,7,8,9,10,11,12,13,14,15);
}
// B-operand fragment: 32x16 bf16. Lane = column; lanes 0-15 hold K={0..7,16..23},
// lanes 16-31 hold K={8..15,24..31}.
__device__ __forceinline__ v16bf frag_b(const unsigned short* lds, int col, int lane) {
  const int ko = (lane < 16) ? 0 : 8;
  v8bf lo = *(const v8bf*)(lds + col * LDT + ko);
  v8bf hi = *(const v8bf*)(lds + col * LDT + 16 + ko);
  return __builtin_shufflevector(lo, hi, 0,1,2,3,4,5,6,7,8,9,10,11,12,13,14,15);
}

// ---------------- kernel 0: x fp32 -> bf16 workspace ----------------
__global__ __launch_bounds__(256)
void cvt_bf16_kernel(const float* __restrict__ x, unsigned short* __restrict__ xb) {
  const size_t i = ((size_t)blockIdx.x * 256 + threadIdx.x) * 8;
  float4 f0 = *(const float4*)(x + i);
  float4 f1 = *(const float4*)(x + i + 4);
  union { unsigned short us[8]; uint4 u4; } pk;
  pk.us[0] = f2bf(f0.x); pk.us[1] = f2bf(f0.y); pk.us[2] = f2bf(f0.z); pk.us[3] = f2bf(f0.w);
  pk.us[4] = f2bf(f1.x); pk.us[5] = f2bf(f1.y); pk.us[6] = f2bf(f1.z); pk.us[7] = f2bf(f1.w);
  *(uint4*)(xb + i) = pk.u4;
}

// ---------------- kernel 1: logits = X * X^T (per batch) ----------------
__global__ __launch_bounds__(256)
void gemm_xxT_kernel(const unsigned short* __restrict__ Xb, float* __restrict__ logits) {
  __shared__ __align__(16) unsigned short lA[2][128 * LDT];
  __shared__ __align__(16) unsigned short lB[2][128 * LDT];
  const int b  = blockIdx.z;
  const int i0 = blockIdx.y * 128, j0 = blockIdx.x * 128;
  const int tid = threadIdx.x, lane = tid & 31, wid = tid >> 5;
  const int wm = (wid >> 2) * 64, wn = (wid & 3) * 32;   // wave tile 64x32
  const unsigned short* Xbb = Xb + (size_t)b * CC * HW;
  const int lrow = tid >> 1, lch = (tid & 1) * 16;
  v8f acc[4][2] = {};

  const unsigned short* ga0 = Xbb + (size_t)(i0 + lrow) * HW + lch;
  const unsigned short* gb0 = Xbb + (size_t)(j0 + lrow) * HW + lch;
  unsigned laoff[2], lboff[2];
  laoff[0] = lds_off(&lA[0][lrow * LDT + lch]); laoff[1] = lds_off(&lA[1][lrow * LDT + lch]);
  lboff[0] = lds_off(&lB[0][lrow * LDT + lch]); lboff[1] = lds_off(&lB[1][lrow * LDT + lch]);

  // prologue: stage first K-tile into buffer 0
  async_b128(laoff[0],      ga0);
  async_b128(laoff[0] + 16, ga0 + 8);
  async_b128(lboff[0],      gb0);
  async_b128(lboff[0] + 16, gb0 + 8);
  wait_async0();
  __syncthreads();

  for (int kb = 0; kb < HW; kb += KB) {
    const int cur = (kb >> 5) & 1;
    if (kb + KB < HW) {   // stage next K-tile into the other buffer (overlaps WMMA)
      const int nxt = cur ^ 1;
      async_b128(laoff[nxt],      ga0 + kb + KB);
      async_b128(laoff[nxt] + 16, ga0 + kb + KB + 8);
      async_b128(lboff[nxt],      gb0 + kb + KB);
      async_b128(lboff[nxt] + 16, gb0 + kb + KB + 8);
    }

    v16bf af[4], bfr[2];
#pragma unroll
    for (int mi = 0; mi < 4; ++mi) af[mi]  = frag_a(lA[cur], wm + mi * 16 + (lane & 15), lane);
#pragma unroll
    for (int ni = 0; ni < 2; ++ni) bfr[ni] = frag_b(lB[cur], wn + ni * 16 + (lane & 15), lane);
#pragma unroll
    for (int mi = 0; mi < 4; ++mi)
#pragma unroll
      for (int ni = 0; ni < 2; ++ni)
        acc[mi][ni] = __builtin_amdgcn_wmma_f32_16x16x32_bf16(
            false, af[mi], false, bfr[ni], (short)0, acc[mi][ni], false, false);

    wait_async0();        // my async writes into nxt have landed
    __syncthreads();      // everyone done reading cur & writing nxt
  }

  float* Lg = logits + (size_t)b * CC * CC;
  const int mo = (lane < 16) ? 0 : 8, nn = lane & 15;
#pragma unroll
  for (int mi = 0; mi < 4; ++mi)
#pragma unroll
    for (int ni = 0; ni < 2; ++ni)
#pragma unroll
      for (int r = 0; r < 8; ++r)
        Lg[(size_t)(i0 + wm + mi * 16 + mo + r) * CC + (j0 + wn + ni * 16 + nn)] =
            acc[mi][ni][r];
}

// ---------------- kernel 2: row softmax + convert A to bf16 ----------------
__global__ __launch_bounds__(256)
void softmax_kernel(const float* __restrict__ logits, unsigned short* __restrict__ Ab) {
  const int lane = threadIdx.x & 31, wid = threadIdx.x >> 5;
  const size_t row = (size_t)blockIdx.x * 8 + wid;   // one wave per 512-wide row
  const float* rp = logits + row * CC;
  float v[16];
  float m = -3.4e38f;
#pragma unroll
  for (int j = 0; j < 16; ++j) { v[j] = rp[lane + j * 32]; m = fmaxf(m, v[j]); }
#pragma unroll
  for (int off = 16; off >= 1; off >>= 1) m = fmaxf(m, __shfl_xor(m, off, 32));
  float s = 0.f;
#pragma unroll
  for (int j = 0; j < 16; ++j) { v[j] = __expf(v[j] - m); s += v[j]; }
#pragma unroll
  for (int off = 16; off >= 1; off >>= 1) s += __shfl_xor(s, off, 32);
  const float inv = 1.0f / s;
  unsigned short* op = Ab + row * CC;
#pragma unroll
  for (int j = 0; j < 16; ++j) op[lane + j * 32] = f2bf(v[j] * inv);
}

// ---------------- kernel 3: out = x + scale * (A * X) ----------------
__global__ __launch_bounds__(256)
void gemm_ax_kernel(const unsigned short* __restrict__ Ab,
                    const unsigned short* __restrict__ Xb,
                    const float* __restrict__ x, const float* __restrict__ scaleP,
                    float* __restrict__ out) {
  __shared__ __align__(16) unsigned short lA[2][128 * LDT];
  __shared__ __align__(16) unsigned short lB[2][128 * LDT];   // transposed [n][k]
  const int b  = blockIdx.z;
  const int m0 = blockIdx.y * 128, n0 = blockIdx.x * 128;
  const int tid = threadIdx.x, lane = tid & 31, wid = tid >> 5;
  const int wm = (wid >> 2) * 64, wn = (wid & 3) * 32;
  const unsigned short* Abb = Ab + (size_t)b * CC * CC;
  const unsigned short* Xbb = Xb + (size_t)b * CC * HW;
  const int arow = tid >> 1, ach = (tid & 1) * 16;   // A tile: 128 rows x 32
  const int bk   = tid >> 3, bch = (tid & 7) * 16;   // B tile: 32 rows x 128
  v8f acc[4][2] = {};

  const unsigned short* ga0 = Abb + (size_t)(m0 + arow) * CC + ach;
  const unsigned short* gbrow = Xbb + (size_t)bk * HW + n0 + bch;
  unsigned laoff[2];
  laoff[0] = lds_off(&lA[0][arow * LDT + ach]); laoff[1] = lds_off(&lA[1][arow * LDT + ach]);

  // manual transpose stage of B tile (bf16 row-major K x N -> LDS [n][k])
  auto stageB = [&](int buf, int kb) {
    const uint4* pb = (const uint4*)(gbrow + (size_t)kb * HW);
    union { uint4 u4[2]; unsigned short us[16]; } pk;
    pk.u4[0] = pb[0]; pk.u4[1] = pb[1];
#pragma unroll
    for (int e = 0; e < 16; ++e) lB[buf][(bch + e) * LDT + bk] = pk.us[e];
  };

  // prologue
  async_b128(laoff[0],      ga0);
  async_b128(laoff[0] + 16, ga0 + 8);
  stageB(0, 0);
  wait_async0();
  __syncthreads();

  for (int kb = 0; kb < CC; kb += KB) {
    const int cur = (kb >> 5) & 1;
    if (kb + KB < CC) {
      const int nxt = cur ^ 1;
      async_b128(laoff[nxt],      ga0 + kb + KB);
      async_b128(laoff[nxt] + 16, ga0 + kb + KB + 8);
      stageB(nxt, kb + KB);
    }

    v16bf af[4], bfr[2];
#pragma unroll
    for (int mi = 0; mi < 4; ++mi) af[mi]  = frag_a(lA[cur], wm + mi * 16 + (lane & 15), lane);
#pragma unroll
    for (int ni = 0; ni < 2; ++ni) bfr[ni] = frag_b(lB[cur], wn + ni * 16 + (lane & 15), lane);
#pragma unroll
    for (int mi = 0; mi < 4; ++mi)
#pragma unroll
      for (int ni = 0; ni < 2; ++ni)
        acc[mi][ni] = __builtin_amdgcn_wmma_f32_16x16x32_bf16(
            false, af[mi], false, bfr[ni], (short)0, acc[mi][ni], false, false);

    wait_async0();
    __syncthreads();
  }

  const float s = scaleP[0];
  const float* xbb = x   + (size_t)b * CC * HW;
  float*       obb = out + (size_t)b * CC * HW;
  const int mo = (lane < 16) ? 0 : 8, nn = lane & 15;
#pragma unroll
  for (int mi = 0; mi < 4; ++mi)
#pragma unroll
    for (int ni = 0; ni < 2; ++ni)
#pragma unroll
      for (int r = 0; r < 8; ++r) {
        const size_t idx = (size_t)(m0 + wm + mi * 16 + mo + r) * HW +
                           (n0 + wn + ni * 16 + nn);
        obb[idx] = xbb[idx] + s * acc[mi][ni][r];
      }
}

extern "C" void kernel_launch(void* const* d_in, const int* in_sizes, int n_in,
                              void* d_out, int out_size, void* d_ws, size_t ws_size,
                              hipStream_t stream) {
  const float* x     = (const float*)d_in[0];
  const float* scale = (const float*)d_in[1];
  float* out = (float*)d_out;
  char* ws = (char*)d_ws;
  // ws layout: [0)            Xb   bf16 64*512*2304  = 150,994,944 B
  //            [150994944)    logits f32 64*512*512  =  67,108,864 B
  //            [218103808)    A    bf16 64*512*512   =  33,554,432 B
  unsigned short* Xb     = (unsigned short*)ws;
  float*          logits = (float*)(ws + 150994944ull);
  unsigned short* Asm    = (unsigned short*)(ws + 218103808ull);

  cvt_bf16_kernel<<<36864, 256, 0, stream>>>(x, Xb);
  dim3 gA(CC / 128, CC / 128, NB);
  gemm_xxT_kernel<<<gA, 256, 0, stream>>>(Xb, logits);
  softmax_kernel<<<(NB * CC) / 8, 256, 0, stream>>>(logits, Asm);
  dim3 gC(HW / 128, CC / 128, NB);
  gemm_ax_kernel<<<gC, 256, 0, stream>>>(Asm, Xb, x, scale, out);
}
